// GraphTransformer_70360154243436
// MI455X (gfx1250) — matrix operand order repaired and checked by reading, not compiled, not run on hardware
//
#include <hip/hip_runtime.h>
#include <cstdint>
#include <cstddef>
#include <math.h>

// ---------------------------------------------------------------------------
// GraphTransformer (2x TransformerConv + mean-pool + FC) for gfx1250 / MI455X
//   - all dense linears run as f16 WMMA (v_wmma_f32_16x16x32_f16),
//     4 independent accumulator chains to avoid WMMA->WMMA hazard NOPs
//   - weight tiles staged to LDS via Tensor Data Mover (tensor_load_to_lds);
//     an opaque never-taken copy branch keeps the LDS stores visible to the
//     compiler so the ds_load B-fragment reads are not folded to undef
//   - edge softmax/aggregation: wave-per-edge gather/scatter with f32 atomics
// ---------------------------------------------------------------------------

#define TC_IN   128
#define TC_HID  256
#define TC_H    4
#define TC_OUT  128

typedef __attribute__((ext_vector_type(8)))  _Float16     v8h;
typedef __attribute__((ext_vector_type(16))) _Float16     v16h;
typedef __attribute__((ext_vector_type(8)))  float        v8f;
typedef __attribute__((ext_vector_type(4)))  unsigned int v4u;
typedef __attribute__((ext_vector_type(8)))  int          v8i;
typedef __attribute__((ext_vector_type(4)))  int          v4i;

#if defined(__HIP_DEVICE_COMPILE__) && __has_builtin(__builtin_amdgcn_tensor_load_to_lds)
#define HAVE_TDM 1
#else
#define HAVE_TDM 0
#endif

// ------------------------------ small helpers ------------------------------

__device__ inline void atomicMaxF(float* addr, float v) {
    unsigned int* ua = (unsigned int*)addr;
    unsigned int old = *ua;
    while (__uint_as_float(old) < v) {
        unsigned int assumed = old;
        old = atomicCAS(ua, assumed, __float_as_uint(v));
        if (old == assumed) break;
    }
}

__global__ void fill_f32(float* __restrict__ p, float v, long long n) {
    long long i = (long long)blockIdx.x * blockDim.x + threadIdx.x;
    if (i < n) p[i] = v;
}

__global__ void f32_to_f16(const float* __restrict__ in, _Float16* __restrict__ out,
                           long long n) {
    long long i = (long long)blockIdx.x * blockDim.x + threadIdx.x;
    if (i < n) out[i] = (_Float16)in[i];
}

// W [K x Ncol] f32 row-major  ->  Wt [Ncol x K] f16 row-major (transposed)
__global__ void w_to_f16t(const float* __restrict__ W, _Float16* __restrict__ Wt,
                          int K, int Ncol) {
    long long idx = (long long)blockIdx.x * blockDim.x + threadIdx.x;
    if (idx < (long long)K * Ncol) {
        int kk = (int)(idx / Ncol);
        int nn = (int)(idx % Ncol);
        Wt[(size_t)nn * K + kk] = (_Float16)W[idx];
    }
}

__global__ void relu_cvt(float* __restrict__ h, _Float16* __restrict__ hh, long long n) {
    long long i = (long long)blockIdx.x * blockDim.x + threadIdx.x;
    if (i < n) {
        float v = fmaxf(h[i], 0.0f);
        h[i] = v;
        hh[i] = (_Float16)v;
    }
}

// ------------------------------ WMMA GEMM ----------------------------------
// C[M x Ncol] = A[M x K](f16) * Bt[Ncol x K](f16, pre-transposed) + bias
// Block: 256 threads = 8 waves; block owns a 16-col tile x 128 rows.
// The 16 x K B-tile (contiguous 32*K bytes of Bt) is staged in LDS via TDM.

__device__ __forceinline__ v16h cat8h(v8h lo, v8h hi) {
    return __builtin_shufflevector(lo, hi,
        0,1,2,3,4,5,6,7,8,9,10,11,12,13,14,15);
}

__device__ __forceinline__ v16h load_a_frag(const _Float16* arow, int kb, int g) {
    // A fragment (16x32 f16): lane(m) holds K = kb+8g..+7 and kb+16+8g..+7
    v8h lo = *(const v8h*)(arow + kb + 8 * g);
    v8h hi = *(const v8h*)(arow + kb + 16 + 8 * g);
    return cat8h(lo, hi);
}

__device__ __forceinline__ v16h load_b_frag(const _Float16* brow, int kb, int g) {
    // B fragment (32x16 f16): lane(n) holds K = kb+8g..+7 and kb+16+8g..+7 (LDS)
    v8h lo = *(const v8h*)(brow + kb + 8 * g);
    v8h hi = *(const v8h*)(brow + kb + 16 + 8 * g);
    return cat8h(lo, hi);
}

__global__ __launch_bounds__(256) void gemm_bias_f16w(
    const _Float16* __restrict__ A, const _Float16* __restrict__ Bt,
    const float* __restrict__ bias, float* __restrict__ C,
    int M, int K, int Ncol, int relu)
{
    __shared__ __align__(16) _Float16 ldsB[16 * 256];   // up to K=256

    const int tiles_col = Ncol >> 4;
    const int col_tile  = blockIdx.x % tiles_col;
    const int row_block = blockIdx.x / tiles_col;
    const int lane = threadIdx.x & 31;
    const int wave = threadIdx.x >> 5;
    const int col0 = col_tile << 4;
    const int row0 = row_block * 128 + wave * 16;

    // opaque zero: prevents the compiler from proving the copy branch dead,
    // so ldsB has a visible writer and LDS reads are not folded to undef.
    int opaque_zero;
    asm volatile("s_mov_b32 %0, 0" : "=s"(opaque_zero));

#if HAVE_TDM
    if (wave == 0) {
        // 1-D contiguous tile: 16*K f16 elements starting at Bt + col0*K -> LDS 0
        const unsigned long long ga = (unsigned long long)(uintptr_t)(Bt + (size_t)col0 * K);
        const unsigned int nelem = 16u * (unsigned int)K;   // in data_size units
        v4u g0;
        g0[0] = 1u;                                         // count=1, user mode
        g0[1] = 0u;                                         // lds_addr = 0 (ldsB base)
        g0[2] = (unsigned int)(ga & 0xffffffffu);           // global_addr[31:0]
        g0[3] = (unsigned int)((ga >> 32) & 0x01ffffffu)    // global_addr[56:32]
              | (2u << 30);                                 // type = 2 (image)
        v8i g1;
        g1[0] = (int)(1u << 16);                            // data_size = 2 bytes
        g1[1] = (int)((nelem & 0xffffu) << 16);             // tensor_dim0[15:0]
        g1[2] = (int)((nelem >> 16) | (1u << 16));          // tensor_dim0[31:16] | tensor_dim1=1
        g1[3] = (int)((nelem & 0xffffu) << 16);             // tile_dim0 = nelem
        g1[4] = 1;                                          // tile_dim1 = 1
        g1[5] = (int)nelem;                                 // tensor_dim0_stride[31:0]
        g1[6] = 0;
        g1[7] = 0;
        v4i gz4 = {0, 0, 0, 0};
        v8i gz8 = {0, 0, 0, 0, 0, 0, 0, 0};
        __builtin_amdgcn_tensor_load_to_lds(g0, g1, gz4, gz4, gz8, 0);
        __builtin_amdgcn_s_wait_tensorcnt(0);
    }
    if (opaque_zero) {   // never taken at runtime (TDM already staged the tile)
        const v8h* srcp = (const v8h*)(Bt + (size_t)col0 * K);
        v8h* dstp = (v8h*)ldsB;
        const int chunks = (16 * K) >> 3;
        for (int i = threadIdx.x; i < chunks; i += blockDim.x) dstp[i] = srcp[i];
    }
#else
    (void)opaque_zero;
    {
        const v8h* srcp = (const v8h*)(Bt + (size_t)col0 * K);
        v8h* dstp = (v8h*)ldsB;
        const int chunks = (16 * K) >> 3;
        for (int i = threadIdx.x; i < chunks; i += blockDim.x) dstp[i] = srcp[i];
    }
#endif
    __syncthreads();

    if (row0 < M) {
        const int nl = lane & 15;
        const int g  = lane >> 4;
        const _Float16* arow = A + (size_t)(row0 + nl) * K;
        const _Float16* brow = ldsB + nl * K;

        // 4 independent accumulator chains -> consecutive WMMAs independent,
        // hazard slots filled with loads instead of v_nop.
        v8f acc0, acc1, acc2, acc3;
#pragma unroll
        for (int r = 0; r < 8; ++r) { acc0[r] = 0.f; acc1[r] = 0.f; acc2[r] = 0.f; acc3[r] = 0.f; }

        int kb = 0;
        for (; kb + 128 <= K; kb += 128) {
            v16h a0 = load_a_frag(arow, kb +  0, g), b0 = load_b_frag(brow, kb +  0, g);
            v16h a1 = load_a_frag(arow, kb + 32, g), b1 = load_b_frag(brow, kb + 32, g);
            v16h a2 = load_a_frag(arow, kb + 64, g), b2 = load_b_frag(brow, kb + 64, g);
            v16h a3 = load_a_frag(arow, kb + 96, g), b3 = load_b_frag(brow, kb + 96, g);
            acc0 = __builtin_amdgcn_wmma_f32_16x16x32_f16(false, a0, false, b0, (short)0, acc0, false, false);
            acc1 = __builtin_amdgcn_wmma_f32_16x16x32_f16(false, a1, false, b1, (short)0, acc1, false, false);
            acc2 = __builtin_amdgcn_wmma_f32_16x16x32_f16(false, a2, false, b2, (short)0, acc2, false, false);
            acc3 = __builtin_amdgcn_wmma_f32_16x16x32_f16(false, a3, false, b3, (short)0, acc3, false, false);
        }
        for (; kb < K; kb += 32) {   // tail (unused for K=128/256, kept general)
            v16h a0 = load_a_frag(arow, kb, g), b0 = load_b_frag(brow, kb, g);
            acc0 = __builtin_amdgcn_wmma_f32_16x16x32_f16(false, a0, false, b0, (short)0, acc0, false, false);
        }

        const float bv = bias[col0 + nl];
        float* ccol = C + (size_t)col0 + nl;
#pragma unroll
        for (int r = 0; r < 8; ++r) {
            float vr = (acc0[r] + acc1[r]) + (acc2[r] + acc3[r]) + bv;
            if (relu) vr = fmaxf(vr, 0.0f);
            ccol[(size_t)(row0 + r + 8 * g) * Ncol] = vr;   // C[row][col0+nl]
        }
    }
}

// ------------------------- edge (attention) kernels ------------------------
// wave32 per edge; lane owns 8 contiguous channels; head = lane>>3.

__global__ void edge_logits(const float* __restrict__ q, const float* __restrict__ k,
                            const int* __restrict__ src, const int* __restrict__ dst,
                            float* __restrict__ logits, float* __restrict__ mbuf, int E)
{
    int e = blockIdx.x * (blockDim.x >> 5) + (threadIdx.x >> 5);
    if (e >= E) return;
    int lane = threadIdx.x & 31;
    int s = src[e], d = dst[e];
    const float* qp = q + (size_t)d * TC_HID + lane * 8;
    const float* kp = k + (size_t)s * TC_HID + lane * 8;
    float part = 0.0f;
#pragma unroll
    for (int i = 0; i < 8; ++i) part += qp[i] * kp[i];
    part += __shfl_xor(part, 1, 32);
    part += __shfl_xor(part, 2, 32);
    part += __shfl_xor(part, 4, 32);          // each 8-lane head group reduced
    if ((lane & 7) == 0) {
        int h = lane >> 3;
        float lg = part * 0.125f;             // 1/sqrt(D=64)
        logits[(size_t)e * TC_H + h] = lg;
        atomicMaxF(&mbuf[(size_t)d * TC_H + h], lg);
    }
}

__global__ void edge_exp(float* __restrict__ logits, const float* __restrict__ mbuf,
                         float* __restrict__ zbuf, const int* __restrict__ dst, int E)
{
    long long idx = (long long)blockIdx.x * blockDim.x + threadIdx.x;
    if (idx >= (long long)E * TC_H) return;
    int e = (int)(idx >> 2);
    int h = (int)(idx & 3);
    int d = dst[e];
    float p = __expf(logits[idx] - mbuf[(size_t)d * TC_H + h]);
    logits[idx] = p;
    atomicAdd(&zbuf[(size_t)d * TC_H + h], p);
}

__global__ void edge_scatter(const float* __restrict__ logits, const float* __restrict__ zbuf,
                             const float* __restrict__ v, const int* __restrict__ src,
                             const int* __restrict__ dst, float* __restrict__ out, int E)
{
    int e = blockIdx.x * (blockDim.x >> 5) + (threadIdx.x >> 5);
    if (e >= E) return;
    int lane = threadIdx.x & 31;
    int s = src[e], d = dst[e];
    int h = lane >> 3;
    float alpha = logits[(size_t)e * TC_H + h]
                / (zbuf[(size_t)d * TC_H + h] + 1e-16f);
    const float* vp = v + (size_t)s * TC_HID + lane * 8;
    float* op = out + (size_t)d * TC_HID + lane * 8;
#pragma unroll
    for (int i = 0; i < 8; ++i) atomicAdd(&op[i], alpha * vp[i]);
}

// ------------------------------- pooling -----------------------------------

__global__ void pool_accum(const float* __restrict__ h, const int* __restrict__ batch,
                           float* __restrict__ pooled, float* __restrict__ cnt, int N)
{
    int node = blockIdx.x * (blockDim.x >> 5) + (threadIdx.x >> 5);
    if (node >= N) return;
    int lane = threadIdx.x & 31;
    int b = batch[node];
    const float* hp = h + (size_t)node * TC_HID + lane * 8;
    float* pp = pooled + (size_t)b * TC_HID + lane * 8;
#pragma unroll
    for (int i = 0; i < 8; ++i) atomicAdd(&pp[i], hp[i]);
    if (lane == 0) atomicAdd(&cnt[b], 1.0f);
}

__global__ void pool_div(float* __restrict__ pooled, const float* __restrict__ cnt,
                         _Float16* __restrict__ pooledh, int G)
{
    int idx = blockIdx.x * blockDim.x + threadIdx.x;
    if (idx < G * TC_HID) {
        float c = fmaxf(cnt[idx / TC_HID], 1.0f);
        float v = pooled[idx] / c;
        pooled[idx] = v;
        pooledh[idx] = (_Float16)v;
    }
}

// ------------------------------- launcher ----------------------------------

extern "C" void kernel_launch(void* const* d_in, const int* in_sizes, int n_in,
                              void* d_out, int out_size, void* d_ws, size_t ws_size,
                              hipStream_t stream) {
    (void)n_in; (void)ws_size;

    const float* x     = (const float*)d_in[0];
    const int*   ei    = (const int*)d_in[1];
    const int*   batch = (const int*)d_in[2];
    const float* Wsrc[9] = {
        (const float*)d_in[3],  (const float*)d_in[5],  (const float*)d_in[7],
        (const float*)d_in[9],  (const float*)d_in[11], (const float*)d_in[13],
        (const float*)d_in[15], (const float*)d_in[17], (const float*)d_in[19] };
    const float* b1q = (const float*)d_in[4];
    const float* b1k = (const float*)d_in[6];
    const float* b1v = (const float*)d_in[8];
    const float* b1s = (const float*)d_in[10];
    const float* b2q = (const float*)d_in[12];
    const float* b2k = (const float*)d_in[14];
    const float* b2v = (const float*)d_in[16];
    const float* b2s = (const float*)d_in[18];
    const float* bfc = (const float*)d_in[20];

    const int N = in_sizes[0] / TC_IN;
    const int E = in_sizes[1] / 2;
    const int G = out_size / TC_OUT;
    const int* src = ei;
    const int* dst = ei + E;

    // ---- carve workspace ----
    uint8_t* wsp = (uint8_t*)d_ws;
    auto carve = [&](size_t bytes) -> void* {
        void* p = (void*)wsp;
        wsp += (bytes + 255) & ~((size_t)255);
        return p;
    };
    _Float16* Xh = (_Float16*)carve((size_t)N * TC_IN * 2);
    const int Wk[9] = {128,128,128,128,256,256,256,256,256};
    const int Wn[9] = {256,256,256,256,256,256,256,256,128};
    _Float16* Wt[9];
    for (int i = 0; i < 9; ++i) Wt[i] = (_Float16*)carve((size_t)Wk[i] * Wn[i] * 2);
    float* qb  = (float*)carve((size_t)N * TC_HID * 4);
    float* kb  = (float*)carve((size_t)N * TC_HID * 4);
    float* vb  = (float*)carve((size_t)N * TC_HID * 4);
    float* hb  = (float*)carve((size_t)N * TC_HID * 4);
    _Float16* hh = (_Float16*)carve((size_t)N * TC_HID * 2);
    float* lg  = (float*)carve((size_t)E * TC_H * 4);
    float* mb  = (float*)carve((size_t)N * TC_H * 4);
    float* zb  = (float*)carve((size_t)N * TC_H * 4);
    float* pooled = (float*)carve((size_t)G * TC_HID * 4);
    float* cnt    = (float*)carve((size_t)G * 4);
    _Float16* pooledh = (_Float16*)carve((size_t)G * TC_HID * 2);

    const int TB = 256;
    auto cdiv = [](long long a, long long b) -> int { return (int)((a + b - 1) / b); };
    auto gemm = [&](const _Float16* A, const _Float16* Bt, const float* bias,
                    float* C, int M, int K, int Ncol, int relu) {
        int grid = ((M + 127) / 128) * (Ncol / 16);
        gemm_bias_f16w<<<grid, 256, 0, stream>>>(A, Bt, bias, C, M, K, Ncol, relu);
    };

    // ---- precision conversion ----
    f32_to_f16<<<cdiv((long long)N * TC_IN, TB), TB, 0, stream>>>(x, Xh, (long long)N * TC_IN);
    for (int i = 0; i < 9; ++i) {
        long long nel = (long long)Wk[i] * Wn[i];
        w_to_f16t<<<cdiv(nel, TB), TB, 0, stream>>>(Wsrc[i], Wt[i], Wk[i], Wn[i]);
    }

    const int EW  = cdiv(E, 8);                 // wave-per-edge kernels
    const long long NH = (long long)N * TC_HID;

    // ======================= layer 1 =======================
    gemm(Xh, Wt[0], b1q, qb, N, TC_IN, TC_HID, 0);
    gemm(Xh, Wt[1], b1k, kb, N, TC_IN, TC_HID, 0);
    gemm(Xh, Wt[2], b1v, vb, N, TC_IN, TC_HID, 0);
    gemm(Xh, Wt[3], b1s, hb, N, TC_IN, TC_HID, 0);   // skip term = aggregation base
    fill_f32<<<cdiv((long long)N * TC_H, TB), TB, 0, stream>>>(mb, -__builtin_huge_valf(), (long long)N * TC_H);
    fill_f32<<<cdiv((long long)N * TC_H, TB), TB, 0, stream>>>(zb, 0.0f, (long long)N * TC_H);
    edge_logits<<<EW, 256, 0, stream>>>(qb, kb, src, dst, lg, mb, E);
    edge_exp<<<cdiv((long long)E * TC_H, TB), TB, 0, stream>>>(lg, mb, zb, dst, E);
    edge_scatter<<<EW, 256, 0, stream>>>(lg, zb, vb, src, dst, hb, E);
    relu_cvt<<<cdiv(NH, TB), TB, 0, stream>>>(hb, hh, NH);

    // ======================= layer 2 =======================
    gemm(hh, Wt[4], b2q, qb, N, TC_HID, TC_HID, 0);
    gemm(hh, Wt[5], b2k, kb, N, TC_HID, TC_HID, 0);
    gemm(hh, Wt[6], b2v, vb, N, TC_HID, TC_HID, 0);
    gemm(hh, Wt[7], b2s, hb, N, TC_HID, TC_HID, 0);  // overwrites h1 (inputs read from hh)
    fill_f32<<<cdiv((long long)N * TC_H, TB), TB, 0, stream>>>(mb, -__builtin_huge_valf(), (long long)N * TC_H);
    fill_f32<<<cdiv((long long)N * TC_H, TB), TB, 0, stream>>>(zb, 0.0f, (long long)N * TC_H);
    edge_logits<<<EW, 256, 0, stream>>>(qb, kb, src, dst, lg, mb, E);
    edge_exp<<<cdiv((long long)E * TC_H, TB), TB, 0, stream>>>(lg, mb, zb, dst, E);
    edge_scatter<<<EW, 256, 0, stream>>>(lg, zb, vb, src, dst, hb, E);
    relu_cvt<<<cdiv(NH, TB), TB, 0, stream>>>(hb, hh, NH);

    // =================== mean pool + FC ====================
    fill_f32<<<cdiv((long long)G * TC_HID, TB), TB, 0, stream>>>(pooled, 0.0f, (long long)G * TC_HID);
    fill_f32<<<1, TB, 0, stream>>>(cnt, 0.0f, (long long)G);
    pool_accum<<<cdiv(N, 8), 256, 0, stream>>>(hb, batch, pooled, cnt, N);
    pool_div<<<cdiv((long long)G * TC_HID, TB), TB, 0, stream>>>(pooled, cnt, pooledh, G);
    gemm(pooledh, Wt[8], bfc, (float*)d_out, G, TC_HID, TC_OUT, 0);
}
